// PAPETransformerEncoderLayer_79482664779897
// MI455X (gfx1250) — matrix operand order repaired and checked by reading, not compiled, not run on hardware
//
#include <hip/hip_runtime.h>

// ---------------------------------------------------------------------------
// PAPE Transformer encoder layer for MI455X (gfx1250), wave32 + WMMA bf16
// + TDM (tensor_load_to_lds) staging of GEMM A-tiles.
// ---------------------------------------------------------------------------

#define BATCH 2
#define SEQ   2048
#define DM    1024
#define NH    16
#define DH    64
#define DFF   4096
#define MTOT  (BATCH * SEQ)   // 4096 rows

typedef __attribute__((ext_vector_type(16))) __bf16 v16bf;
typedef __attribute__((ext_vector_type(8)))  __bf16 v8bf;
typedef __attribute__((ext_vector_type(8)))  float  v8f;
typedef __attribute__((ext_vector_type(4)))  unsigned int u32x4;
typedef __attribute__((ext_vector_type(8)))  unsigned int u32x8;

union AF { v16bf v; v8bf h[2]; };

#define WMMA_BF16(a, b, c) \
    __builtin_amdgcn_wmma_f32_16x16x32_bf16(false, (a), false, (b), (short)0, (c), false, false)

// ---------------------------------------------------------------------------
// TDM: issue a 2D bf16 tile load (tile1 rows x tile0 elems) from global to
// LDS via the Tensor Data Mover.  Descriptor packing per CDNA5 ISA 8.3/8.4:
//   group0: [1:0]=count=1 | [63:32]=lds_addr | [120:64]=global_addr | [127:126]=2
//   group1: [17:16]=data_size(1=2B) | [79:48]=tensor_dim0 | [111:80]=tensor_dim1
//           | [127:112]=tile_dim0 | [143:128]=tile_dim1 | [207:160]=dim0_stride
// Issued per-wave (EXEC ignored); wave-uniform inputs live in SGPRs.
// ---------------------------------------------------------------------------
__device__ __forceinline__ void tdm_load_2d_bf16(unsigned lds_off, const void* gaddr,
                                                 unsigned dim0, unsigned dim1,
                                                 unsigned tile0, unsigned tile1,
                                                 unsigned stride0) {
    unsigned long long ga = (unsigned long long)gaddr;
    u32x4 g0;
    g0[0] = 1u;                                    // count=1 (valid user descriptor)
    g0[1] = lds_off;                               // lds_addr
    g0[2] = (unsigned)ga;                          // global_addr[31:0]
    g0[3] = (unsigned)(ga >> 32) | (2u << 30);     // global_addr[56:32] | type=2
    u32x8 g1;
    g1[0] = 1u << 16;                              // workgroup_mask=0, data_size=1 (2B)
    g1[1] = (dim0 & 0xFFFFu) << 16;                // tensor_dim0[15:0]
    g1[2] = (dim0 >> 16) | ((dim1 & 0xFFFFu) << 16);   // dim0[31:16] | dim1[15:0]
    g1[3] = (dim1 >> 16) | (tile0 << 16);          // dim1[31:16] | tile_dim0
    g1[4] = tile1 & 0xFFFFu;                       // tile_dim1 (tile_dim2=0)
    g1[5] = stride0;                               // tensor_dim0_stride[31:0]
    g1[6] = 0u;                                    // stride0[47:32], dim1_stride lo
    g1[7] = 0u;
    asm volatile("tensor_load_to_lds %0, %1" :: "s"(g0), "s"(g1) : "memory");
}

// ---------------------------------------------------------------------------
// f32 -> bf16 cast
// ---------------------------------------------------------------------------
__global__ void cast_f32_bf16(const float* __restrict__ in, __bf16* __restrict__ out, int n) {
    int i = blockIdx.x * blockDim.x + threadIdx.x;
    if (i < n) out[i] = (__bf16)in[i];
}

// ---------------------------------------------------------------------------
// V [B,S,D] -> Vt [B,D,S]  (so P*V B-fragments are contiguous in kv)
// ---------------------------------------------------------------------------
__global__ void transpose_v(const __bf16* __restrict__ V, __bf16* __restrict__ Vt) {
    int i = blockIdx.x * blockDim.x + threadIdx.x;     // over BATCH*DM*SEQ
    int s  = i % SEQ;
    int hd = (i / SEQ) % DM;
    int b  = i / (SEQ * DM);
    Vt[i] = V[((size_t)(b * SEQ + s)) * DM + hd];
}

// ---------------------------------------------------------------------------
// Generic GEMM: C[M,N] = A[M,K] * W[N,K]^T + bias, optional exact GELU.
// Block = 8 waves sharing one 32-row A tile (block tile 32 x 512).
// A is staged 32x64 at a time into double-buffered LDS by wave 0 via the
// Tensor Data Mover (tensor_load_to_lds + s_wait_tensorcnt), giving 8x reuse
// of the A stream.  Each wave computes a 32x64 C tile: per 32-wide k-step it
// loads 2 A fragments from LDS + 4 B fragments from global and issues
// 8 x v_wmma_f32_16x16x32_bf16 (B reused across both m-subtiles, halving the
// dominant weight-stream traffic per FLOP vs a 16-row tile).
// Fragment layouts used here (same contiguous k order for A and B):
//   A 16x32: M=lane&15; lane<16 -> K=0..15, lane>=16 -> K=16..31
//   B 32x16: N=lane&15; lane<16 -> K=0..15, lane>=16 -> K=16..31
// ---------------------------------------------------------------------------
__global__ void gemm_bf16_kernel(const __bf16* __restrict__ A, const __bf16* __restrict__ W,
                                 const float* __restrict__ bias,
                                 float* __restrict__ Cf, __bf16* __restrict__ Cb,
                                 int M, int N, int K, int act) {
    __shared__ __align__(128) __bf16 As[2][32][64];   // double-buffered A tile (2 x 4KB)

    const int w    = threadIdx.x >> 5;
    const int lane = threadIdx.x & 31;
    const int nblk = N >> 9;                 // 512-wide n blocks
    const int mt   = blockIdx.x / nblk;
    const int nb   = blockIdx.x - mt * nblk;
    const int m0   = mt << 5;
    const int n0   = (nb << 9) + (w << 6);

    v8f acc[2][4];
    #pragma unroll
    for (int s = 0; s < 2; ++s)
        #pragma unroll
        for (int i = 0; i < 4; ++i)
            #pragma unroll
            for (int r = 0; r < 8; ++r) acc[s][i][r] = 0.0f;

    const int ka = (lane & 16) ? 8 : 0;
    const int kb = (lane & 16) ? 16 : 0;
    const unsigned ldsA[2] = { (unsigned)(unsigned long long)(uintptr_t)&As[0][0][0],
                               (unsigned)(unsigned long long)(uintptr_t)&As[1][0][0] };

    if (w == 0)   // wave-uniform: whole wave issues one TDM op for chunk 0
        tdm_load_2d_bf16(ldsA[0], A + (size_t)m0 * K, (unsigned)K, (unsigned)M,
                         64u, 32u, (unsigned)K);

    for (int kc = 0; kc < K; kc += 64) {
        const int cur = (kc >> 6) & 1;
        if (w == 0) {
            if (kc + 64 < K) {   // prefetch next A chunk, keep 1 TDM in flight
                tdm_load_2d_bf16(ldsA[cur ^ 1], A + (size_t)m0 * K + kc + 64,
                                 (unsigned)K, (unsigned)M, 64u, 32u, (unsigned)K);
                __builtin_amdgcn_s_wait_tensorcnt(1);
            } else {
                __builtin_amdgcn_s_wait_tensorcnt(0);
            }
        }
        __syncthreads();   // A chunk `cur` visible to all 8 waves

        #pragma unroll
        for (int kk = 0; kk < 64; kk += 32) {
            AF af0, af1;
            af0.h[0] = *(const v8bf*)&As[cur][lane & 15][kk + kb];
            af0.h[1] = *(const v8bf*)&As[cur][lane & 15][kk + kb + 8];
            af1.h[0] = *(const v8bf*)&As[cur][16 + (lane & 15)][kk + kb];
            af1.h[1] = *(const v8bf*)&As[cur][16 + (lane & 15)][kk + kb + 8];
            #pragma unroll
            for (int nt = 0; nt < 4; ++nt) {
                const __bf16* rowW =
                    W + (size_t)(n0 + nt * 16 + (lane & 15)) * K + kc + kk + kb;
                v16bf bf = *(const v16bf*)rowW;
                acc[0][nt] = WMMA_BF16(af0.v, bf, acc[0][nt]);
                acc[1][nt] = WMMA_BF16(af1.v, bf, acc[1][nt]);
            }
        }
        __syncthreads();   // all waves done with chunk before it is overwritten
    }

    // C layout: VGPR r -> row + 8*(lane>=16) + r, col = n0 + nt*16 + (lane&15)
    const int rhalf = (lane >> 4) & 1;
    const int col0  = lane & 15;
    #pragma unroll
    for (int s = 0; s < 2; ++s) {
        #pragma unroll
        for (int nt = 0; nt < 4; ++nt) {
            int coln = n0 + nt * 16 + col0;
            float bb = bias ? bias[coln] : 0.0f;
            #pragma unroll
            for (int r = 0; r < 8; ++r) {
                int row = m0 + s * 16 + rhalf * 8 + r;
                float v = acc[s][nt][r] + bb;
                if (act) v = 0.5f * v * (1.0f + erff(v * 0.70710678118f));  // exact GELU
                size_t idx = (size_t)row * N + coln;
                if (Cf) Cf[idx] = v;
                if (Cb) Cb[idx] = (__bf16)v;
            }
        }
    }
}

// ---------------------------------------------------------------------------
// Flash attention with additive relative-position bias.
// One wave per (b, h, 16-row q tile). Streams kv in chunks of 32:
//   scores: 2 WMMAs per 16x16 tile (Dh=64 contraction),
//   + bias, online softmax (shfl row reductions), P -> LDS -> A-fragment,
//   P*V: 4 WMMAs per chunk (Vt gives contiguous B-fragments).
// The 256MB bias stream is prefetched one chunk ahead (global_prefetch_b8).
// ---------------------------------------------------------------------------
__global__ void flash_attn_kernel(const __bf16* __restrict__ Q, const __bf16* __restrict__ Kb,
                                  const __bf16* __restrict__ Vt, const float* __restrict__ rb,
                                  __bf16* __restrict__ O) {
    __shared__ __align__(32) __bf16 lp[8][16][32];   // per-wave P tile (16 x 32)

    int wid  = (blockIdx.x * blockDim.x + threadIdx.x) >> 5;
    int lane = threadIdx.x & 31;
    int w    = threadIdx.x >> 5;

    int qt = wid & (SEQ / 16 - 1);          // 0..127
    int h  = (wid >> 7) & (NH - 1);         // 0..15
    int b  = wid >> 11;                     // 0..1
    int q0 = qt * 16;

    const int ka    = (lane & 16) ? 8 : 0;
    const int kb16  = (lane & 16) ? 16 : 0;
    const int rhalf = (lane >> 4) & 1;
    const int col   = lane & 15;
    const float scale = 0.125f;             // 1/sqrt(64)

    // Q fragments for d=0..31 and d=32..63
    AF qf[2];
    {
        const __bf16* rowQ = Q + (size_t)(b * SEQ + q0 + col) * DM + h * DH;
        qf[0].h[0] = *(const v8bf*)(rowQ + kb16);
        qf[0].h[1] = *(const v8bf*)(rowQ + kb16 + 8);
        qf[1].h[0] = *(const v8bf*)(rowQ + 32 + kb16);
        qf[1].h[1] = *(const v8bf*)(rowQ + 40 + kb16);
    }

    float mi[8], li[8];
    v8f oa[4];
    #pragma unroll
    for (int r = 0; r < 8; ++r) { mi[r] = -1e30f; li[r] = 0.0f; }
    #pragma unroll
    for (int nt = 0; nt < 4; ++nt)
        #pragma unroll
        for (int r = 0; r < 8; ++r) oa[nt][r] = 0.0f;

    for (int kv0 = 0; kv0 < SEQ; kv0 += 32) {
        // prefetch next chunk of the bias stream (one cacheline per q-row)
        if (kv0 + 32 < SEQ)
            __builtin_prefetch(rb + ((size_t)h * SEQ + q0 + col) * SEQ + kv0 + 32, 0, 0);

        // --- scores: two 16x16 tiles over this 32-wide kv chunk ---
        v8f s0, s1;
        #pragma unroll
        for (int r = 0; r < 8; ++r) { s0[r] = 0.0f; s1[r] = 0.0f; }
        {
            const __bf16* rowK0 = Kb + (size_t)(b * SEQ + kv0 + col) * DM + h * DH;
            v16bf b0 = *(const v16bf*)(rowK0 + kb16);
            v16bf b1 = *(const v16bf*)(rowK0 + 32 + kb16);
            s0 = WMMA_BF16(qf[0].v, b0, s0);
            s0 = WMMA_BF16(qf[1].v, b1, s0);
            const __bf16* rowK1 = rowK0 + (size_t)16 * DM;
            v16bf b2 = *(const v16bf*)(rowK1 + kb16);
            v16bf b3 = *(const v16bf*)(rowK1 + 32 + kb16);
            s1 = WMMA_BF16(qf[0].v, b2, s1);
            s1 = WMMA_BF16(qf[1].v, b3, s1);
        }

        // --- bias + online softmax (rows live across 16-lane halves) ---
        #pragma unroll
        for (int r = 0; r < 8; ++r) {
            int qrow = q0 + rhalf * 8 + r;
            const float* brow = rb + ((size_t)h * SEQ + qrow) * SEQ + kv0;
            float a0 = s0[r] * scale + brow[col];
            float a1 = s1[r] * scale + brow[16 + col];
            float mx = fmaxf(a0, a1);
            mx = fmaxf(mx, __shfl_xor(mx, 1));
            mx = fmaxf(mx, __shfl_xor(mx, 2));
            mx = fmaxf(mx, __shfl_xor(mx, 4));
            mx = fmaxf(mx, __shfl_xor(mx, 8));
            float mnew = fmaxf(mi[r], mx);
            float p0 = __expf(a0 - mnew);
            float p1 = __expf(a1 - mnew);
            float corr = __expf(mi[r] - mnew);
            float ps = p0 + p1;
            ps += __shfl_xor(ps, 1);
            ps += __shfl_xor(ps, 2);
            ps += __shfl_xor(ps, 4);
            ps += __shfl_xor(ps, 8);
            li[r] = li[r] * corr + ps;
            mi[r] = mnew;
            oa[0][r] *= corr; oa[1][r] *= corr; oa[2][r] *= corr; oa[3][r] *= corr;
            int trow = rhalf * 8 + r;
            lp[w][trow][col]      = (__bf16)p0;
            lp[w][trow][16 + col] = (__bf16)p1;
        }

        // --- re-swizzle P from C-layout to A-fragment layout via LDS ---
        AF pf;
        pf.h[0] = *(const v8bf*)&lp[w][col][kb16];
        pf.h[1] = *(const v8bf*)&lp[w][col][kb16 + 8];

        // --- P * V : Vt[b][h*64+d][s], contiguous 16 kv per lane ---
        #pragma unroll
        for (int nt = 0; nt < 4; ++nt) {
            const __bf16* rowV =
                Vt + (size_t)(b * DM + h * DH + nt * 16 + col) * SEQ + kv0 + kb16;
            v16bf vf = *(const v16bf*)rowV;
            oa[nt] = WMMA_BF16(pf.v, vf, oa[nt]);
        }
    }

    // --- normalize rows and store bf16 attention output [B*S, DM] ---
    #pragma unroll
    for (int r = 0; r < 8; ++r) {
        float inv = 1.0f / li[r];
        int row = q0 + rhalf * 8 + r;
        size_t base = (size_t)(b * SEQ + row) * DM + h * DH;
        #pragma unroll
        for (int nt = 0; nt < 4; ++nt)
            O[base + nt * 16 + col] = (__bf16)(oa[nt][r] * inv);
    }
}

// ---------------------------------------------------------------------------
// out = LayerNorm(a + b) * gamma + beta ; optional bf16 copy for next GEMM.
// One 256-thread block per 1024-wide row.
// ---------------------------------------------------------------------------
__global__ void add_ln_kernel(const float* __restrict__ a, const float* __restrict__ bsrc,
                              const float* __restrict__ g, const float* __restrict__ be,
                              float* __restrict__ outf, __bf16* __restrict__ outb) {
    __shared__ float red[256];
    int row = blockIdx.x;
    const float* pa = a + (size_t)row * DM;
    const float* pb = bsrc + (size_t)row * DM;

    float v[4];
    float s = 0.0f;
    #pragma unroll
    for (int j = 0; j < 4; ++j) {
        int c = threadIdx.x + j * 256;
        v[j] = pa[c] + pb[c];
        s += v[j];
    }
    red[threadIdx.x] = s;
    __syncthreads();
    for (int off = 128; off > 0; off >>= 1) {
        if (threadIdx.x < off) red[threadIdx.x] += red[threadIdx.x + off];
        __syncthreads();
    }
    float mean = red[0] * (1.0f / DM);
    __syncthreads();

    float s2 = 0.0f;
    #pragma unroll
    for (int j = 0; j < 4; ++j) {
        float d = v[j] - mean;
        s2 += d * d;
    }
    red[threadIdx.x] = s2;
    __syncthreads();
    for (int off = 128; off > 0; off >>= 1) {
        if (threadIdx.x < off) red[threadIdx.x] += red[threadIdx.x + off];
        __syncthreads();
    }
    float rstd = rsqrtf(red[0] * (1.0f / DM) + 1e-5f);

    #pragma unroll
    for (int j = 0; j < 4; ++j) {
        int c = threadIdx.x + j * 256;
        float o = (v[j] - mean) * rstd * g[c] + be[c];
        size_t idx = (size_t)row * DM + c;
        outf[idx] = o;
        if (outb) outb[idx] = (__bf16)o;
    }
}

// ---------------------------------------------------------------------------
// Host launcher
// ---------------------------------------------------------------------------
extern "C" void kernel_launch(void* const* d_in, const int* in_sizes, int n_in,
                              void* d_out, int out_size, void* d_ws, size_t ws_size,
                              hipStream_t stream) {
    const float* src = (const float*)d_in[0];
    const float* rb  = (const float*)d_in[1];
    const float* Wq  = (const float*)d_in[2];   const float* bq = (const float*)d_in[3];
    const float* Wk  = (const float*)d_in[4];   const float* bk = (const float*)d_in[5];
    const float* Wv  = (const float*)d_in[6];   const float* bv = (const float*)d_in[7];
    const float* Wo  = (const float*)d_in[8];   const float* bo = (const float*)d_in[9];
    const float* W1  = (const float*)d_in[10];  const float* b1 = (const float*)d_in[11];
    const float* W2  = (const float*)d_in[12];  const float* b2 = (const float*)d_in[13];
    const float* g1  = (const float*)d_in[14];  const float* be1 = (const float*)d_in[15];
    const float* g2  = (const float*)d_in[16];  const float* be2 = (const float*)d_in[17];
    float* out = (float*)d_out;

    char* p = (char*)d_ws;
    auto alloc = [&](size_t bytes) -> void* {
        void* r = (void*)p;
        p += (bytes + 255) & ~(size_t)255;
        return r;
    };
    __bf16* xbf   = (__bf16*)alloc((size_t)MTOT * DM * 2);
    __bf16* wqb   = (__bf16*)alloc((size_t)DM * DM * 2);
    __bf16* wkb   = (__bf16*)alloc((size_t)DM * DM * 2);
    __bf16* wvb   = (__bf16*)alloc((size_t)DM * DM * 2);
    __bf16* wob   = (__bf16*)alloc((size_t)DM * DM * 2);
    __bf16* w1b   = (__bf16*)alloc((size_t)DFF * DM * 2);
    __bf16* w2b   = (__bf16*)alloc((size_t)DM * DFF * 2);
    __bf16* qb    = (__bf16*)alloc((size_t)MTOT * DM * 2);
    __bf16* kbf   = (__bf16*)alloc((size_t)MTOT * DM * 2);
    __bf16* vb    = (__bf16*)alloc((size_t)MTOT * DM * 2);
    __bf16* vtb   = (__bf16*)alloc((size_t)MTOT * DM * 2);
    __bf16* attnb = (__bf16*)alloc((size_t)MTOT * DM * 2);
    float*  attnf = (float*) alloc((size_t)MTOT * DM * 4);
    float*  x1f   = (float*) alloc((size_t)MTOT * DM * 4);
    __bf16* x1b   = (__bf16*)alloc((size_t)MTOT * DM * 2);
    __bf16* hb    = (__bf16*)alloc((size_t)MTOT * DFF * 2);
    float*  fff   = (float*) alloc((size_t)MTOT * DM * 4);

    auto cast = [&](const float* in, __bf16* o, int n) {
        cast_f32_bf16<<<(n + 255) / 256, 256, 0, stream>>>(in, o, n);
    };
    cast(src, xbf, MTOT * DM);
    cast(Wq, wqb, DM * DM);
    cast(Wk, wkb, DM * DM);
    cast(Wv, wvb, DM * DM);
    cast(Wo, wob, DM * DM);
    cast(W1, w1b, DFF * DM);
    cast(W2, w2b, DM * DFF);

    auto gemm = [&](const __bf16* A, const __bf16* W, const float* bias,
                    float* Cf, __bf16* Cb, int M, int N, int K, int act) {
        int blocks = (M / 32) * (N / 512);   // 8 waves/block, 32x512 block tile
        gemm_bf16_kernel<<<blocks, 256, 0, stream>>>(A, W, bias, Cf, Cb, M, N, K, act);
    };

    // QKV projections (bf16 out)
    gemm(xbf, wqb, bq, nullptr, qb,  MTOT, DM, DM, 0);
    gemm(xbf, wkb, bk, nullptr, kbf, MTOT, DM, DM, 0);
    gemm(xbf, wvb, bv, nullptr, vb,  MTOT, DM, DM, 0);

    // V -> Vt [B, D, S]
    transpose_v<<<(MTOT * DM) / 256, 256, 0, stream>>>(vb, vtb);

    // flash attention: B*H*(S/16) = 4096 waves
    flash_attn_kernel<<<4096 * 32 / 256, 256, 0, stream>>>(qb, kbf, vtb, rb, attnb);

    // output projection (f32 out)
    gemm(attnb, wob, bo, attnf, nullptr, MTOT, DM, DM, 0);

    // x1 = LN(src + attn)
    add_ln_kernel<<<MTOT, 256, 0, stream>>>(src, attnf, g1, be1, x1f, x1b);

    // FFN
    gemm(x1b, w1b, b1, nullptr, hb, MTOT, DFF, DM, 1);   // + exact GELU
    gemm(hb, w2b, b2, fff, nullptr, MTOT, DM, DFF, 0);

    // out = LN(x1 + ff)
    add_ln_kernel<<<MTOT, 256, 0, stream>>>(x1f, fff, g2, be2, out, nullptr);

    (void)in_sizes; (void)n_in; (void)out_size; (void)ws_size;
}
